// BiLSTM_CRF_81088982548816
// MI455X (gfx1250) — matrix operand order repaired and checked by reading, not compiled, not run on hardware
//
#include <hip/hip_runtime.h>
#include <hip/hip_bf16.h>
#include <math.h>

// ---------------- problem constants ----------------
#define Vv   50000
#define Ee   128
#define Hh   256
#define HHc  128
#define Kc   11
#define STARTt 9
#define STOPt  10
#define Bb   128
#define Tt   1024
#define Gg   512          // 4*HH
#define NEGV (-10000.0f)

typedef __bf16 bf16;
typedef __attribute__((ext_vector_type(16))) __bf16 v16bf;
typedef __attribute__((ext_vector_type(4)))  __bf16 v4bf;
typedef __attribute__((ext_vector_type(8)))  float  v8f;

union AFrag { v16bf v; __bf16 e[16]; };
union CFrag { v8f   v; float  e[8];  };
union BPack { v4bf  v; __bf16 e[4];  };

__device__ __forceinline__ float rcp_fast(float x) { return __builtin_amdgcn_rcpf(x); }
__device__ __forceinline__ float sigf(float x)     { return rcp_fast(1.0f + __expf(-x)); }
// tanh(x) = 1 - 2/(1 + e^{2x})
__device__ __forceinline__ float tanh_fast(float x) {
    return fmaf(-2.0f, rcp_fast(1.0f + __expf(2.0f * x)), 1.0f);
}

// Workgroup barrier that drains ONLY DScnt (LDS producer->consumer ordering).
// Unlike __syncthreads(), it does NOT wait for in-flight global loads, so the
// emb-row prefetch issued above barrier (A) keeps flying under the WMMAs.
__device__ __forceinline__ void lds_barrier() {
    asm volatile("s_wait_dscnt 0x0\n\t"
                 "s_barrier_signal -1\n\t"
                 "s_barrier_wait -1" ::: "memory");
}

// =====================================================================
// Kernel 1: fused BiLSTM recurrence.
// grid = 16 blocks : blockIdx = dir*8 + batch_tile ; block = 512 thr (16 waves)
// Per step: gates[16,512] = [x_t | h_{t-1}](16x256) * Wcomb^T(256x512) + bias,
// via v_wmma_f32_16x16x32_bf16.
//  - weight B-fragments: VGPR-resident across all T steps (128 VGPRs/wave)
//  - bias folded into accumulator init; cell state c loop-carried in VGPRs
//  - x_t: LDS double buffer; emb row prefetched 1 step / word 2 steps ahead,
//    branchless (index clamped); DScnt-only barriers keep the prefetch in
//    flight across barrier (A)
//  - A fragments: 4-deep rotated LDS feed (3 fragments in flight under wmma)
// =====================================================================
__global__ __launch_bounds__(512)
void bilstm_kernel(const int* __restrict__ sentence,
                   const float* __restrict__ h0, const float* __restrict__ c0,
                   const float* __restrict__ emb,
                   const float* __restrict__ Wih_f, const float* __restrict__ Whh_f,
                   const float* __restrict__ b_f,
                   const float* __restrict__ Wih_b, const float* __restrict__ Whh_b,
                   const float* __restrict__ b_b,
                   bf16* __restrict__ hf_out, bf16* __restrict__ hb_out)
{
    const int dir   = blockIdx.x >> 3;
    const int b0    = (blockIdx.x & 7) * 16;
    const float* Wih  = dir ? Wih_b : Wih_f;
    const float* Whh  = dir ? Whh_b : Whh_f;
    const float* bias = dir ? b_b   : b_f;
    bf16* hout = dir ? hb_out : hf_out;

    __shared__ __align__(16) bf16  xbuf[2][16][136];  // x_t double buffer (K-tiles 0..3)
    __shared__ __align__(16) bf16  hbuf[16][136];     // h_{t-1}           (K-tiles 4..7)
    __shared__ __align__(16) float gates[16][512];

    const int tid  = threadIdx.x;
    const int wave = tid >> 5;         // 0..15 : gate cols [wave*32, wave*32+32) AND row m=wave
    const int lane = tid & 31;
    const int m    = wave;             // batch row owned for staging / elementwise
    const int j0   = lane * 4;         // 4 consecutive columns per lane

    // bias for this wave-lane's two output-tile columns (same for all 8 rows)
    const float bval0 = bias[wave * 32 + (lane & 15)];
    const float bval1 = bias[wave * 32 + 16 + (lane & 15)];

    // ---- init h (LDS) and c (registers) ----
    float4 cv = *(const float4*)&c0[((size_t)dir * Bb + b0 + m) * HHc + j0];
    {
        const float4 hv = *(const float4*)&h0[((size_t)dir * Bb + b0 + m) * HHc + j0];
        BPack hp;
        hp.e[0] = (bf16)hv.x; hp.e[1] = (bf16)hv.y; hp.e[2] = (bf16)hv.z; hp.e[3] = (bf16)hv.w;
        *(v4bf*)&hbuf[m][j0] = hp.v;
    }

    // ---- preload B fragments (W^T) into registers: 2 N-tiles x 8 K-tiles ----
    AFrag bw[2][8];
    {
        const int n    = lane & 15;
        const int koff = (lane >> 4) * 8;
#pragma unroll
        for (int nt = 0; nt < 2; ++nt) {
            const int g = wave * 32 + nt * 16 + n;   // W row = gate index
#pragma unroll
            for (int kt = 0; kt < 8; ++kt) {
                const float* src = (kt < 4) ? &Wih[g * 128 + kt * 32 + koff]
                                            : &Whh[g * 128 + (kt - 4) * 32 + koff];
#pragma unroll
                for (int i = 0; i < 16; ++i) bw[nt][kt].e[i] = (bf16)src[(i < 8) ? i : (8 + i)];
            }
        }
    }

    // ---- software pipeline prologue: stage x for t=0, word for t=1 ----
    const int* sent_row = sentence + (size_t)(b0 + m) * Tt;
    {
        const int tt0 = dir ? (Tt - 1) : 0;
        const int w0  = sent_row[tt0];
        const float4 xf = *(const float4*)&emb[(size_t)w0 * Ee + j0];
        BPack xp;
        xp.e[0] = (bf16)xf.x; xp.e[1] = (bf16)xf.y; xp.e[2] = (bf16)xf.z; xp.e[3] = (bf16)xf.w;
        *(v4bf*)&xbuf[0][m][j0] = xp.v;
    }
    int word_next = sent_row[dir ? (Tt - 2) : 1];   // word for t=1

    const int arow  = lane & 15;
    const int abase = (lane >> 4) * 8;
    const int cn    = lane & 15;
    const int cmh   = (lane >> 4) * 8;

    for (int t = 0; t < Tt; ++t) {
        const int tt  = dir ? (Tt - 1 - t) : t;
        const int cur = t & 1, nxt = cur ^ 1;

        // ---- issue prefetches for t+1 / t+2 (branchless; stays in flight
        //      across barrier (A) since barriers drain DScnt only) ----
        const float4 xf_n = *(const float4*)&emb[(size_t)word_next * Ee + j0];
        const int tcl = (t + 2 < Tt) ? (t + 2) : (Tt - 1);      // clamped
        const int word_n2 = (t + 2 < Tt) ? sent_row[dir ? (Tt - 1 - tcl) : tcl] : 0;

        lds_barrier();     // (A) xbuf[cur] + hbuf ready; prev gates consumed

        // ---- GEMM: 2 output tiles per wave, K = 256, A 4-deep pipelined ----
        CFrag acc0, acc1;
#pragma unroll
        for (int r = 0; r < 8; ++r) { acc0.e[r] = bval0; acc1.e[r] = bval1; }

        AFrag afr[4];
#pragma unroll
        for (int p = 0; p < 3; ++p) {
            const bf16* ap = (p < 4) ? &xbuf[cur][arow][p * 32]
                                     : &hbuf[arow][(p - 4) * 32];
#pragma unroll
            for (int i = 0; i < 8; ++i) afr[p].e[i]     = ap[abase + i];
#pragma unroll
            for (int i = 0; i < 8; ++i) afr[p].e[8 + i] = ap[16 + abase + i];
        }
#pragma unroll
        for (int kt = 0; kt < 8; ++kt) {
            if (kt + 3 < 8) {
                const int kn = kt + 3;
                const bf16* ap = (kn < 4) ? &xbuf[cur][arow][kn * 32]
                                          : &hbuf[arow][(kn - 4) * 32];
                AFrag& an = afr[kn & 3];
#pragma unroll
                for (int i = 0; i < 8; ++i) an.e[i]     = ap[abase + i];
#pragma unroll
                for (int i = 0; i < 8; ++i) an.e[8 + i] = ap[16 + abase + i];
            }
            acc0.v = __builtin_amdgcn_wmma_f32_16x16x32_bf16(
                false, afr[kt & 3].v, false, bw[0][kt].v, (short)0, acc0.v, false, false);
            acc1.v = __builtin_amdgcn_wmma_f32_16x16x32_bf16(
                false, afr[kt & 3].v, false, bw[1][kt].v, (short)0, acc1.v, false, false);
        }
#pragma unroll
        for (int r = 0; r < 8; ++r) {
            gates[cmh + r][wave * 32 + cn]      = acc0.e[r];
            gates[cmh + r][wave * 32 + 16 + cn] = acc1.e[r];
        }

        // keep the prefetch drain (cvt + loadcnt wait) below the matrix work
        __builtin_amdgcn_sched_barrier(0);

        // ---- drain prefetched x_{t+1} into the other buffer ----
        {
            BPack xp;
            xp.e[0] = (bf16)xf_n.x; xp.e[1] = (bf16)xf_n.y;
            xp.e[2] = (bf16)xf_n.z; xp.e[3] = (bf16)xf_n.w;
            *(v4bf*)&xbuf[nxt][m][j0] = xp.v;
        }
        word_next = word_n2;

        lds_barrier();     // (B) gates + xbuf[nxt] visible

        // ---- gate nonlinearities, c/h update (wave w -> row m=w, float4) ----
        {
            const float4 iv = *(const float4*)&gates[m][j0];
            const float4 fv = *(const float4*)&gates[m][128 + j0];
            const float4 gv = *(const float4*)&gates[m][256 + j0];
            const float4 ov = *(const float4*)&gates[m][384 + j0];
            float h[4];
            {
                float c = sigf(fv.x) * cv.x + sigf(iv.x) * tanh_fast(gv.x);
                h[0] = sigf(ov.x) * tanh_fast(c); cv.x = c;
                c = sigf(fv.y) * cv.y + sigf(iv.y) * tanh_fast(gv.y);
                h[1] = sigf(ov.y) * tanh_fast(c); cv.y = c;
                c = sigf(fv.z) * cv.z + sigf(iv.z) * tanh_fast(gv.z);
                h[2] = sigf(ov.z) * tanh_fast(c); cv.z = c;
                c = sigf(fv.w) * cv.w + sigf(iv.w) * tanh_fast(gv.w);
                h[3] = sigf(ov.w) * tanh_fast(c); cv.w = c;
            }
            BPack hp;
            hp.e[0] = (bf16)h[0]; hp.e[1] = (bf16)h[1];
            hp.e[2] = (bf16)h[2]; hp.e[3] = (bf16)h[3];
            *(v4bf*)&hbuf[m][j0] = hp.v;
            *(v4bf*)&hout[((size_t)(b0 + m) * Tt + tt) * HHc + j0] = hp.v;
        }
    }
}

// =====================================================================
// Kernel 2: feats = [hf|hb] @ W_out^T + b_out  (N padded 11 -> 16), WMMA.
// grid = B*T/64 blocks, 4 waves each; each wave owns 16 rows of BT.
// =====================================================================
__global__ __launch_bounds__(128)
void feats_kernel(const bf16* __restrict__ hf, const bf16* __restrict__ hb,
                  const float* __restrict__ W_out, const float* __restrict__ b_out,
                  float* __restrict__ feats)
{
    const int lane = threadIdx.x & 31;
    const int wave = threadIdx.x >> 5;
    const int r0   = blockIdx.x * 64 + wave * 16;
    const int n    = lane & 15;
    const int koff = (lane >> 4) * 8;

    AFrag bw[8];
#pragma unroll
    for (int kt = 0; kt < 8; ++kt)
#pragma unroll
        for (int i = 0; i < 16; ++i) {
            int k = kt * 32 + koff + ((i < 8) ? i : (8 + i));
            bw[kt].e[i] = (n < Kc) ? (bf16)W_out[n * Hh + k] : (bf16)0.0f;
        }

    CFrag acc;
    acc.v = (v8f){0.f, 0.f, 0.f, 0.f, 0.f, 0.f, 0.f, 0.f};
    const int mm  = lane & 15;
    const int ab  = (lane >> 4) * 8;
    const size_t row = (size_t)(r0 + mm);
#pragma unroll
    for (int kt = 0; kt < 8; ++kt) {
        const bf16* src = (kt < 4) ? &hf[row * 128 + kt * 32]
                                   : &hb[row * 128 + (kt - 4) * 32];
        AFrag a;
#pragma unroll
        for (int i = 0; i < 8; ++i) a.e[i]     = src[ab + i];
#pragma unroll
        for (int i = 0; i < 8; ++i) a.e[8 + i] = src[16 + ab + i];
        acc.v = __builtin_amdgcn_wmma_f32_16x16x32_bf16(
            false, a.v, false, bw[kt].v, (short)0, acc.v, false, false);
    }

    const int mh = (lane >> 4) * 8;
    if (n < Kc) {
        float bo = b_out[n];
#pragma unroll
        for (int r = 0; r < 8; ++r)
            feats[(size_t)(r0 + mh + r) * Kc + n] = acc.e[r] + bo;
    }
}

// =====================================================================
// Kernel 3: CRF forward scan + gold score, one wave per batch element.
// Lanes 0..10 = next-tag; alpha broadcast by shuffle.
// =====================================================================
__global__ __launch_bounds__(32)
void crf_kernel(const float* __restrict__ feats, const int* __restrict__ tags,
                const float* __restrict__ trans, float* __restrict__ scores)
{
    const int b    = blockIdx.x;
    const int lane = threadIdx.x;

    float tr[Kc];
#pragma unroll
    for (int j = 0; j < Kc; ++j) tr[j] = (lane < Kc) ? trans[lane * Kc + j] : NEGV;

    float alpha = (lane == STARTt) ? 0.0f : NEGV;

    for (int t = 0; t < Tt; ++t) {
        float feat = (lane < Kc) ? feats[((size_t)b * Tt + t) * Kc + lane] : 0.0f;
        float v[Kc];
        float best = -3.0e38f;
#pragma unroll
        for (int j = 0; j < Kc; ++j) {
            float aj = __shfl(alpha, j, 32);
            v[j] = aj + tr[j];
            best = fmaxf(best, v[j]);
        }
        float s = 0.0f;
#pragma unroll
        for (int j = 0; j < Kc; ++j) s += __expf(v[j] - best);
        float na = best + __logf(s) + feat;
        alpha = (lane < Kc) ? na : NEGV;
    }

    // fwd = logsumexp(alpha + trans[STOP, :])
    float term = (lane < Kc) ? (alpha + trans[STOPt * Kc + lane]) : -3.0e38f;
    float mx = term;
#pragma unroll
    for (int off = 16; off > 0; off >>= 1) mx = fmaxf(mx, __shfl_xor(mx, off, 32));
    float ex = (lane < Kc) ? __expf(term - mx) : 0.0f;
#pragma unroll
    for (int off = 16; off > 0; off >>= 1) ex += __shfl_xor(ex, off, 32);
    float fwd = mx + __logf(ex);

    // gold path score (lane-partitioned over t)
    float g = 0.0f;
    for (int t = lane; t < Tt; t += 32) {
        int cur  = tags[(size_t)b * Tt + t];
        int prev = (t == 0) ? STARTt : tags[(size_t)b * Tt + t - 1];
        g += trans[cur * Kc + prev] + feats[((size_t)b * Tt + t) * Kc + cur];
    }
#pragma unroll
    for (int off = 16; off > 0; off >>= 1) g += __shfl_xor(g, off, 32);

    if (lane == 0) {
        g += trans[STOPt * Kc + tags[(size_t)b * Tt + Tt - 1]];
        scores[b] = fwd - g;
    }
}

// =====================================================================
// Kernel 4: deterministic tree reduction of 128 per-batch scores.
// =====================================================================
__global__ __launch_bounds__(128)
void reduce_kernel(const float* __restrict__ scores, float* __restrict__ out)
{
    __shared__ float sm[128];
    sm[threadIdx.x] = scores[threadIdx.x];
    __syncthreads();
    for (int s = 64; s > 0; s >>= 1) {
        if (threadIdx.x < s) sm[threadIdx.x] += sm[threadIdx.x + s];
        __syncthreads();
    }
    if (threadIdx.x == 0) out[0] = sm[0];
}

// =====================================================================
extern "C" void kernel_launch(void* const* d_in, const int* in_sizes, int n_in,
                              void* d_out, int out_size, void* d_ws, size_t ws_size,
                              hipStream_t stream)
{
    const int*   sentence = (const int*)  d_in[0];
    const int*   tags     = (const int*)  d_in[1];
    const float* h0       = (const float*)d_in[2];
    const float* c0       = (const float*)d_in[3];
    const float* emb      = (const float*)d_in[4];
    const float* W_ih_f   = (const float*)d_in[5];
    const float* W_hh_f   = (const float*)d_in[6];
    const float* b_f      = (const float*)d_in[7];
    const float* W_ih_b   = (const float*)d_in[8];
    const float* W_hh_b   = (const float*)d_in[9];
    const float* b_b      = (const float*)d_in[10];
    const float* W_out    = (const float*)d_in[11];
    const float* b_out    = (const float*)d_in[12];
    const float* trans    = (const float*)d_in[13];
    float* out = (float*)d_out;

    // workspace layout
    char* ws = (char*)d_ws;
    const size_t n_h = (size_t)Bb * Tt * HHc;        // 16,777,216
    bf16*  hf     = (bf16*)ws;                         ws += n_h * sizeof(bf16);
    bf16*  hb     = (bf16*)ws;                         ws += n_h * sizeof(bf16);
    float* feats  = (float*)ws;                        ws += (size_t)Bb * Tt * Kc * sizeof(float);
    float* scores = (float*)ws;

    bilstm_kernel<<<16, 512, 0, stream>>>(sentence, h0, c0, emb,
                                          W_ih_f, W_hh_f, b_f,
                                          W_ih_b, W_hh_b, b_b,
                                          hf, hb);
    feats_kernel<<<(Bb * Tt) / 64, 128, 0, stream>>>(hf, hb, W_out, b_out, feats);
    crf_kernel<<<Bb, 32, 0, stream>>>(feats, tags, trans, scores);
    reduce_kernel<<<1, 128, 0, stream>>>(scores, out);
}
